// RobustFill_47055661694992
// MI455X (gfx1250) — matrix operand order repaired and checked by reading, not compiled, not run on hardware
//
#include <hip/hip_runtime.h>
#include <math.h>

// Problem constants
#define NTOK 512
#define S 1024          // string/embedding size (input dim of in/out LSTMs)
#define H 2048          // program size (hidden dim)
#define G 8192          // 4*H gate rows
#define LSEQ 128        // input/output sequence length (129 steps incl. EOS)
#define PROG 64         // decoder steps

typedef __attribute__((ext_vector_type(16))) __bf16 v16bf;
typedef __attribute__((ext_vector_type(8)))  __bf16 v8bf;
typedef __attribute__((ext_vector_type(8)))  float  v8f;

// ---------------------------------------------------------------- utilities

__global__ void cvt_f32_bf16(const float* __restrict__ src,
                             __bf16* __restrict__ dst, long n) {
  long i = (long)blockIdx.x * blockDim.x + threadIdx.x;
  long stride = (long)gridDim.x * blockDim.x;
  for (; i < n; i += stride) dst[i] = (__bf16)src[i];
}

// Gather token embeddings (incl. EOS at row t==L) and convert to bf16.
__global__ void gather_emb_bf16(const int* __restrict__ toks,
                                const float* __restrict__ emb,
                                __bf16* __restrict__ dst, int L) {
  int t = blockIdx.x;                      // 0..L
  int tok = (t < L) ? toks[t] : NTOK;
  const float* row = emb + (size_t)tok * S;
  __bf16* d = dst + (size_t)t * S;
  for (int j = threadIdx.x; j < S; j += blockDim.x) d[j] = (__bf16)row[j];
}

__global__ void init_state(float* __restrict__ h, float* __restrict__ c,
                           __bf16* __restrict__ hb0, __bf16* __restrict__ hb1) {
  int i = blockIdx.x * blockDim.x + threadIdx.x;
  if (i < H) {
    h[i] = 0.f; c[i] = 0.f;
    hb0[i] = (__bf16)0.f; hb1[i] = (__bf16)0.f;
  }
}

__device__ __forceinline__ float sigf(float x) { return 1.f / (1.f + __expf(-x)); }

// One 16x16x32 bf16 WMMA accumulation step of a batch-1 matvec.
// A tile: rows rowPtr[.. +15] (per-lane row = lane&15), K = k0..k0+31.
// B tile: vector vec broadcast across all 16 columns.
__device__ __forceinline__ v8f wmma_mv(v8f acc,
                                       const __bf16* __restrict__ rowPtr,
                                       const __bf16* __restrict__ vec,
                                       int k0, int halfSel) {
  // A (16x32 bf16): lane<16 holds K{0..15}, lane>=16 holds K{16..31}
  v16bf a = *(const v16bf*)(rowPtr + k0 + (halfSel << 4));
  // B (32x16 bf16) broadcast: lane group selects K{0..15} vs K{16..31}
  v16bf b = *(const v16bf*)(vec + k0 + (halfSel << 4));
  return __builtin_amdgcn_wmma_f32_16x16x32_bf16(false, a, false, b,
                                                 (short)0, acc, false, false);
}

// --------------------------------------------------------------- LSTM step
//
// gates[G] = Wih[G,KX] @ x[KX] + Whh[G,H] @ hin[H]  (+ biases in pointwise)
// 128 blocks x 256 threads (8 wave32s). Wave w: gate quarter (w&3), K-slice
// (w>>2) of both matvecs -> 1024 independent WMMA streams per step; K-slice
// partials are summed in LDS before the elementwise gate update.
template<int KX>
__global__ __launch_bounds__(256)
void lstm_step(const __bf16* __restrict__ Wih, const __bf16* __restrict__ Whh,
               const float* __restrict__ bih, const float* __restrict__ bhh,
               const __bf16* __restrict__ x,
               const __bf16* __restrict__ hin,
               float* __restrict__ h, float* __restrict__ c,
               __bf16* __restrict__ hout) {
  __shared__ float gl[2][4][16];

  const int tid     = threadIdx.x;
  const int wv      = tid >> 5;
  const int quarter = wv & 3;          // gate quarter (i,f,g,o)
  const int kslice  = wv >> 2;         // 0: low K half, 1: high K half
  const int lane    = tid & 31;
  const int halfSel = lane >> 4;       // 0: lanes 0-15, 1: lanes 16-31
  const int m       = lane & 15;       // row within 16-row tile
  const int rowBase = quarter * H + blockIdx.x * 16;

  const __bf16* __restrict__ rowIh =
      Wih + (size_t)(rowBase + m) * KX + kslice * (KX / 2);
  const __bf16* __restrict__ rowHh =
      Whh + (size_t)(rowBase + m) * H + kslice * (H / 2);
  const __bf16* __restrict__ xk = x   + kslice * (KX / 2);
  const __bf16* __restrict__ hk = hin + kslice * (H / 2);

  v8f acc = {};

  // ---- Wih @ x (this wave's K half) ----
  #pragma unroll 4
  for (int k0 = 0; k0 < KX / 2; k0 += 32)
    acc = wmma_mv(acc, rowIh, xk, k0, halfSel);

  // ---- Whh @ h_prev (this wave's K half) ----
  #pragma unroll 4
  for (int k0 = 0; k0 < H / 2; k0 += 32)
    acc = wmma_mv(acc, rowHh, hk, k0, halfSel);

  // D layout: acc[r] = row r (lanes 0-15) / row 8+r (lanes 16-31); all
  // columns identical (B broadcast) -> lanes 0 and 16 publish the 16 rows.
  if (m == 0) {
    #pragma unroll
    for (int r = 0; r < 8; ++r) gl[kslice][quarter][halfSel * 8 + r] = acc[r];
  }
  __syncthreads();

  if (tid < 16) {
    int j = blockIdx.x * 16 + tid;     // hidden index within quarter
    float iv = gl[0][0][tid] + gl[1][0][tid] + bih[j]         + bhh[j];
    float fv = gl[0][1][tid] + gl[1][1][tid] + bih[H + j]     + bhh[H + j];
    float gv = gl[0][2][tid] + gl[1][2][tid] + bih[2 * H + j] + bhh[2 * H + j];
    float ov = gl[0][3][tid] + gl[1][3][tid] + bih[3 * H + j] + bhh[3 * H + j];
    float cn = sigf(fv) * c[j] + sigf(iv) * tanhf(gv);
    float hn = sigf(ov) * tanhf(cn);
    c[j] = cn;
    h[j] = hn;
    hout[j] = (__bf16)hn;
  }
}

// ------------------------------------------------------------- log-softmax
__global__ __launch_bounds__(256)
void log_softmax_2048(const float* __restrict__ h, float* __restrict__ out) {
  __shared__ float red[256];
  int t = threadIdx.x;

  float m = -INFINITY;
  for (int j = t; j < H; j += 256) m = fmaxf(m, h[j]);
  red[t] = m; __syncthreads();
  for (int s = 128; s > 0; s >>= 1) {
    if (t < s) red[t] = fmaxf(red[t], red[t + s]);
    __syncthreads();
  }
  m = red[0]; __syncthreads();

  float sum = 0.f;
  for (int j = t; j < H; j += 256) sum += __expf(h[j] - m);
  red[t] = sum; __syncthreads();
  for (int s = 128; s > 0; s >>= 1) {
    if (t < s) red[t] += red[t + s];
    __syncthreads();
  }
  float lse = m + __logf(red[0]);
  for (int j = t; j < H; j += 256) out[j] = h[j] - lse;
}

// ------------------------------------------------------------------- host

extern "C" void kernel_launch(void* const* d_in, const int* in_sizes, int n_in,
                              void* d_out, int out_size, void* d_ws, size_t ws_size,
                              hipStream_t stream) {
  const int*   in_seq  = (const int*)  d_in[0];
  const int*   out_seq = (const int*)  d_in[1];
  const float* emb     = (const float*)d_in[2];
  const float* WihIn   = (const float*)d_in[3];
  const float* WhhIn   = (const float*)d_in[4];
  const float* bihIn   = (const float*)d_in[5];
  const float* bhhIn   = (const float*)d_in[6];
  const float* WihOut  = (const float*)d_in[7];
  const float* WhhOut  = (const float*)d_in[8];
  const float* bihOut  = (const float*)d_in[9];
  const float* bhhOut  = (const float*)d_in[10];
  const float* WihPr   = (const float*)d_in[11];
  const float* WhhPr   = (const float*)d_in[12];
  const float* bihPr   = (const float*)d_in[13];
  const float* bhhPr   = (const float*)d_in[14];
  float* out = (float*)d_out;

  // -------- workspace layout (256B aligned); total ~168.5 MB --------
  char* base = (char*)d_ws;
  size_t off = 0;
  auto alloc = [&](size_t bytes) -> char* {
    char* p = base + off;
    off += (bytes + 255) & ~(size_t)255;
    return p;
  };
  __bf16* wIhIn  = (__bf16*)alloc((size_t)G * S * 2);
  __bf16* wHhIn  = (__bf16*)alloc((size_t)G * H * 2);
  __bf16* wIhOut = (__bf16*)alloc((size_t)G * S * 2);
  __bf16* wHhOut = (__bf16*)alloc((size_t)G * H * 2);
  __bf16* wIhPr  = (__bf16*)alloc((size_t)G * H * 2);
  __bf16* wHhPr  = (__bf16*)alloc((size_t)G * H * 2);
  __bf16* inEmb  = (__bf16*)alloc((size_t)(LSEQ + 1) * S * 2);
  __bf16* outEmb = (__bf16*)alloc((size_t)(LSEQ + 1) * S * 2);
  __bf16* hb0    = (__bf16*)alloc((size_t)H * 2);
  __bf16* hb1    = (__bf16*)alloc((size_t)H * 2);
  float*  hF     = (float*) alloc((size_t)H * 4);
  float*  cF     = (float*) alloc((size_t)H * 4);
  (void)ws_size; (void)n_in; (void)in_sizes; (void)out_size;

  // -------- per-call prep: bf16 weights, embeddings, state --------
  cvt_f32_bf16<<<4096, 256, 0, stream>>>(WihIn,  wIhIn,  (long)G * S);
  cvt_f32_bf16<<<8192, 256, 0, stream>>>(WhhIn,  wHhIn,  (long)G * H);
  cvt_f32_bf16<<<4096, 256, 0, stream>>>(WihOut, wIhOut, (long)G * S);
  cvt_f32_bf16<<<8192, 256, 0, stream>>>(WhhOut, wHhOut, (long)G * H);
  cvt_f32_bf16<<<8192, 256, 0, stream>>>(WihPr,  wIhPr,  (long)G * H);
  cvt_f32_bf16<<<8192, 256, 0, stream>>>(WhhPr,  wHhPr,  (long)G * H);
  gather_emb_bf16<<<LSEQ + 1, 256, 0, stream>>>(in_seq,  emb, inEmb,  LSEQ);
  gather_emb_bf16<<<LSEQ + 1, 256, 0, stream>>>(out_seq, emb, outEmb, LSEQ);
  init_state<<<(H + 255) / 256, 256, 0, stream>>>(hF, cF, hb0, hb1);

  __bf16* hb[2] = {hb0, hb1};
  int par = 0;
  const int nblk = G / (4 * 16);   // 128 blocks, 16 hidden rows each

  // -------- encode input string --------
  for (int t = 0; t <= LSEQ; ++t) {
    lstm_step<S><<<nblk, 256, 0, stream>>>(
        wIhIn, wHhIn, bihIn, bhhIn,
        inEmb + (size_t)t * S, hb[par], hF, cF, hb[par ^ 1]);
    par ^= 1;
  }
  // -------- encode output string (shared h,c) --------
  for (int t = 0; t <= LSEQ; ++t) {
    lstm_step<S><<<nblk, 256, 0, stream>>>(
        wIhOut, wHhOut, bihOut, bhhOut,
        outEmb + (size_t)t * S, hb[par], hF, cF, hb[par ^ 1]);
    par ^= 1;
  }
  // -------- decode: x = h_prev, emit log_softmax(h_new) --------
  for (int t = 0; t < PROG; ++t) {
    lstm_step<H><<<nblk, 256, 0, stream>>>(
        wIhPr, wHhPr, bihPr, bhhPr,
        hb[par], hb[par], hF, cF, hb[par ^ 1]);
    par ^= 1;
    log_softmax_2048<<<1, 256, 0, stream>>>(hF, out + (size_t)t * H);
  }
}
